// LSTMCell_38560216383688
// MI455X (gfx1250) — compile-verified, matrix operand order checked by reading, not run on hardware
//
#include <hip/hip_runtime.h>
#include <hip/hip_bf16.h>

typedef __attribute__((ext_vector_type(16))) __bf16 v16bf;
typedef __attribute__((ext_vector_type(8)))  float  v8f;
typedef __attribute__((ext_vector_type(4)))  float  v4f;

#define BATCH 256
#define HDIM  1024
#define KDIM  1024
#define NDIM  4096   // 4*H

#define TILE_BYTES (128 * 32 * 4)   // one fp32 stage buffer: 16 KB

// ---------------------------------------------------------------------------
// Load one 16x32 WMMA fragment from an fp32 LDS tile (row-major, ld=32) and
// split into bf16 hi/lo on the fly.
// ISA layout (16-bit A 16x32): lanes 0-15 hold K=0..7 and 16..23 of row
// M=lane; lanes 16-31 hold K=8..15 and 24..31 of row M=lane-16.
// ---------------------------------------------------------------------------
__device__ __forceinline__ void frag_ld_split(const float* base, int lane,
                                              v16bf& hi, v16bf& lo) {
    const float* p = base + (lane & 15) * 32 + ((lane >> 4) << 3);
    v4f q0 = *(const v4f*)(p);
    v4f q1 = *(const v4f*)(p + 4);
    v4f q2 = *(const v4f*)(p + 16);
    v4f q3 = *(const v4f*)(p + 20);
    float x[16];
#pragma unroll
    for (int j = 0; j < 4; ++j) {
        x[j] = q0[j]; x[4 + j] = q1[j]; x[8 + j] = q2[j]; x[12 + j] = q3[j];
    }
#pragma unroll
    for (int j = 0; j < 16; ++j) {
        __bf16 h = (__bf16)x[j];
        hi[j] = h;
        lo[j] = (__bf16)(x[j] - (float)h);
    }
}

// ---------------------------------------------------------------------------
// Issue one stage of async global->LDS DMA: 128x32 fp32 tiles of X and W.
// 256 threads, 4 iterations x 2 matrices = 8 async issues per wave;
// each instruction moves 32 lanes x 16B.
// ---------------------------------------------------------------------------
__device__ __forceinline__ void stage_async(unsigned ldsA, unsigned ldsB,
                                            const float* __restrict__ X,
                                            const float* __restrict__ W,
                                            int row0, int col0, int k0, int t) {
#pragma unroll
    for (int i = 0; i < 4; ++i) {
        const int idx = t + i * 256;       // 0..1023 -> 128 rows x 8 quads
        const int r   = idx >> 3;
        const int kq  = (idx & 7) << 2;    // 0,4,...,28
        const float* ga = X + (size_t)(row0 + r) * KDIM + k0 + kq;
        const float* gb = W + (size_t)(col0 + r) * KDIM + k0 + kq;
        const unsigned la = ldsA + (unsigned)((r * 32 + kq) * 4);
        const unsigned lb = ldsB + (unsigned)((r * 32 + kq) * 4);
        asm volatile("global_load_async_to_lds_b128 %0, %1, off"
                     :: "v"(la), "v"(ga) : "memory");
        asm volatile("global_load_async_to_lds_b128 %0, %1, off"
                     :: "v"(lb), "v"(gb) : "memory");
    }
}

// ---------------------------------------------------------------------------
// GEMM: Out[256 x 4096] = X[256 x 1024] * W[4096 x 1024]^T
// bf16x3 split (hi/lo) on v_wmma_f32_16x16x32_bf16 for ~fp32 accuracy.
// Async double-buffered LDS staging (GLOBAL_LOAD_ASYNC_TO_LDS_B128).
// Block: 256 thr (8 waves), tile 128x128, K-step 32. Wave tile 64x32.
// grid = (NDIM/128, BATCH/128, 2); z picks (inp,W_i)->ig or (hx,W_h)->hg.
// ---------------------------------------------------------------------------
__global__ __launch_bounds__(256)
void lnlstm_gates_gemm(const float* __restrict__ inp, const float* __restrict__ hx,
                       const float* __restrict__ Wi,  const float* __restrict__ Wh,
                       float* __restrict__ ig, float* __restrict__ hg) {
    __shared__ float sA[2][128 * 32];   // 2 x 16 KB
    __shared__ float sB[2][128 * 32];   // 2 x 16 KB

    const float* X   = (blockIdx.z == 0) ? inp : hx;
    const float* W   = (blockIdx.z == 0) ? Wi  : Wh;
    float*       Out = (blockIdx.z == 0) ? ig  : hg;

    const int t    = threadIdx.x;
    const int lane = t & 31;
    const int wv   = t >> 5;       // wave 0..7
    const int wm   = wv >> 2;      // 0..1  (wave row  -> 64 rows)
    const int wn   = wv & 3;       // 0..3  (wave col  -> 32 cols)
    const int row0 = blockIdx.y * 128;
    const int col0 = blockIdx.x * 128;

    const unsigned baseA = (unsigned)(uintptr_t)&sA[0][0];
    const unsigned baseB = (unsigned)(uintptr_t)&sB[0][0];

    v8f acc[4][2];
#pragma unroll
    for (int mt = 0; mt < 4; ++mt)
#pragma unroll
        for (int nt = 0; nt < 2; ++nt) acc[mt][nt] = (v8f)0.0f;

    // Prologue: kick off stage 0 into buffer 0.
    stage_async(baseA, baseB, X, W, row0, col0, 0, t);

    for (int k0 = 0; k0 < KDIM; k0 += 32) {
        const int cur = (k0 >> 5) & 1;
        if (k0 + 32 < KDIM) {
            // Prefetch next stage into the other buffer, then retire only the
            // 8 older issues (async loads complete in order).
            stage_async(baseA + (cur ^ 1) * TILE_BYTES,
                        baseB + (cur ^ 1) * TILE_BYTES,
                        X, W, row0, col0, k0 + 32, t);
            asm volatile("s_wait_asynccnt 0x8" ::: "memory");
        } else {
            asm volatile("s_wait_asynccnt 0x0" ::: "memory");
        }
        __syncthreads();   // current buffer complete across all waves

        const float* tileA = &sA[cur][0];
        const float* tileB = &sB[cur][0];

        v16bf Ahi[4], Alo[4], Bhi[2], Blo[2];
#pragma unroll
        for (int mt = 0; mt < 4; ++mt)
            frag_ld_split(tileA + (wm * 64 + mt * 16) * 32, lane, Ahi[mt], Alo[mt]);
#pragma unroll
        for (int nt = 0; nt < 2; ++nt)
            frag_ld_split(tileB + (wn * 32 + nt * 16) * 32, lane, Bhi[nt], Blo[nt]);

#pragma unroll
        for (int mt = 0; mt < 4; ++mt)
#pragma unroll
            for (int nt = 0; nt < 2; ++nt) {
                v8f c = acc[mt][nt];
                c = __builtin_amdgcn_wmma_f32_16x16x32_bf16(false, Ahi[mt], false, Bhi[nt], (short)0, c, false, false);
                c = __builtin_amdgcn_wmma_f32_16x16x32_bf16(false, Ahi[mt], false, Blo[nt], (short)0, c, false, false);
                c = __builtin_amdgcn_wmma_f32_16x16x32_bf16(false, Alo[mt], false, Bhi[nt], (short)0, c, false, false);
                acc[mt][nt] = c;
            }

        __syncthreads();   // all waves done reading `cur` before it is reused
    }

    // C/D layout: lanes 0-15 -> N=lane, M=v; lanes 16-31 -> N=lane-16, M=v+8.
#pragma unroll
    for (int mt = 0; mt < 4; ++mt)
#pragma unroll
        for (int nt = 0; nt < 2; ++nt)
#pragma unroll
            for (int v = 0; v < 8; ++v) {
                const int gr = row0 + wm * 64 + mt * 16 + v + ((lane >> 4) << 3);
                const int gc = col0 + wn * 32 + nt * 16 + (lane & 15);
                Out[(size_t)gr * NDIM + gc] = acc[mt][nt][v];
            }
}

// ---------------------------------------------------------------------------
// Fused epilogue: per-gate LayerNorm (x8), gating, cell update, cy LayerNorm.
// One block (1024 thr) per batch row; reductions via shfl_xor + LDS.
// ---------------------------------------------------------------------------
__device__ __forceinline__ float block_sum(float v, float* red) {
#pragma unroll
    for (int o = 16; o > 0; o >>= 1) v += __shfl_xor(v, o, 32);
    const int lane = threadIdx.x & 31;
    const int w    = threadIdx.x >> 5;
    if (lane == 0) red[w] = v;
    __syncthreads();
    if (w == 0) {
        float s = red[lane];
#pragma unroll
        for (int o = 16; o > 0; o >>= 1) s += __shfl_xor(s, o, 32);
        if (lane == 0) red[0] = s;
    }
    __syncthreads();
    float r = red[0];
    __syncthreads();
    return r;
}

__device__ __forceinline__ float sigmoidf(float x) {
    return 1.0f / (1.0f + __expf(-x));
}

__global__ __launch_bounds__(1024)
void lnlstm_pointwise(const float* __restrict__ ig, const float* __restrict__ hg,
                      const float* __restrict__ cx, const float* __restrict__ bias,
                      const float* __restrict__ gI, const float* __restrict__ bI,
                      const float* __restrict__ gH, const float* __restrict__ bH,
                      const float* __restrict__ gC, const float* __restrict__ bC,
                      float* __restrict__ hy, float* __restrict__ cy) {
    __shared__ float red[32];
    const int b = blockIdx.x;
    const int h = threadIdx.x;
    const float inv = 1.0f / (float)HDIM;
    const float eps = 1e-10f;

    float pre[4];
#pragma unroll
    for (int g = 0; g < 4; ++g) {
        const int gi = g * HDIM + h;
        float xi  = ig[(size_t)b * NDIM + gi];
        float mui = block_sum(xi, red) * inv;
        float di  = xi - mui;
        float vi  = block_sum(di * di, red) * inv;
        float ni  = di * __frsqrt_rn(vi + eps) * gI[gi] + bI[gi];

        float xh  = hg[(size_t)b * NDIM + gi];
        float muh = block_sum(xh, red) * inv;
        float dh  = xh - muh;
        float vh  = block_sum(dh * dh, red) * inv;
        float nh  = dh * __frsqrt_rn(vh + eps) * gH[gi] + bH[gi];

        pre[g] = ni + nh + bias[gi];
    }

    const float in_g  = sigmoidf(pre[0]);
    const float fg_g  = sigmoidf(pre[1]);
    const float cl_g  = tanhf(pre[2]);
    const float out_g = sigmoidf(pre[3]);

    const float cyv = fg_g * cx[(size_t)b * HDIM + h] + in_g * cl_g;
    cy[(size_t)b * HDIM + h] = cyv;

    float muc = block_sum(cyv, red) * inv;
    float dc  = cyv - muc;
    float vc  = block_sum(dc * dc, red) * inv;
    float nc  = dc * __frsqrt_rn(vc + eps) * gC[h] + bC[h];

    hy[(size_t)b * HDIM + h] = out_g * tanhf(nc);
}

// ---------------------------------------------------------------------------
extern "C" void kernel_launch(void* const* d_in, const int* in_sizes, int n_in,
                              void* d_out, int out_size, void* d_ws, size_t ws_size,
                              hipStream_t stream) {
    (void)in_sizes; (void)n_in; (void)out_size; (void)ws_size;
    const float* inp  = (const float*)d_in[0];
    const float* hx   = (const float*)d_in[1];
    const float* cx   = (const float*)d_in[2];
    const float* W_i  = (const float*)d_in[3];
    const float* W_h  = (const float*)d_in[4];
    const float* bias = (const float*)d_in[5];
    const float* gI   = (const float*)d_in[6];
    const float* bI   = (const float*)d_in[7];
    const float* gH   = (const float*)d_in[8];
    const float* bH   = (const float*)d_in[9];
    const float* gC   = (const float*)d_in[10];
    const float* bC   = (const float*)d_in[11];

    float* hy = (float*)d_out;                      // [256,1024]
    float* cy = hy + (size_t)BATCH * HDIM;          // [256,1024]
    float* ig = (float*)d_ws;                       // [256,4096]
    float* hg = ig + (size_t)BATCH * NDIM;          // [256,4096]

    dim3 ggrid(NDIM / 128, BATCH / 128, 2);
    lnlstm_gates_gemm<<<ggrid, dim3(256), 0, stream>>>(inp, hx, W_i, W_h, ig, hg);

    lnlstm_pointwise<<<dim3(BATCH), dim3(1024), 0, stream>>>(
        ig, hg, cx, bias, gI, bI, gH, bH, gC, bC, hy, cy);
}
